// Superattention_conv_77129022701764
// MI455X (gfx1250) — matrix operand order repaired and checked by reading, not compiled, not run on hardware
//
#include <hip/hip_runtime.h>
#include <hip/hip_bf16.h>

typedef __attribute__((ext_vector_type(16))) __bf16 v16bf;
typedef __attribute__((ext_vector_type(8)))  float  v8f;

#define NEGV (-1e9f)
#define EPSV (1e-5f)

__device__ __forceinline__ unsigned short f2bf_bits(float f) {
    unsigned u = __float_as_uint(f);
    unsigned r = u + 0x7FFFu + ((u >> 16) & 1u);   // round-to-nearest-even bf16
    return (unsigned short)(r >> 16);
}

// ---------------------------------------------------------------------------
// Ragged length reduction: lens = max(label)+1 per batch
// ---------------------------------------------------------------------------
__global__ __launch_bounds__(256) void len_reduce_kernel(
    const int* __restrict__ lmT, const int* __restrict__ lmR,
    int* __restrict__ lt, int* __restrict__ lr, int HW)
{
    int b   = blockIdx.x >> 7;        // 128 blocks per batch
    int blk = blockIdx.x & 127;
    long long base = (long long)b * HW + (long long)blk * 512;
    int tid = threadIdx.x;
    int mt = 0, mr = 0;
    for (int i = tid; i < 512; i += 256) {
        mt = max(mt, lmT[base + i] + 1);
        mr = max(mr, lmR[base + i] + 1);
    }
    for (int o = 16; o > 0; o >>= 1) {
        mt = max(mt, __shfl_down(mt, o, 32));
        mr = max(mr, __shfl_down(mr, o, 32));
    }
    __shared__ int smT[8], smR[8];
    if ((tid & 31) == 0) { smT[tid >> 5] = mt; smR[tid >> 5] = mr; }
    __syncthreads();
    if (tid == 0) {
        int a = 0, c = 0;
        for (int i = 0; i < 8; i++) { a = max(a, smT[i]); c = max(c, smR[i]); }
        atomicMax(&lt[b], a);
        atomicMax(&lr[b], c);
    }
}

// ---------------------------------------------------------------------------
// Center + L2-normalize over channels (optional), transpose [B][C][L] f32 ->
// [B][L][C] bf16. 16 positions per block, LDS-staged.
// ---------------------------------------------------------------------------
__global__ __launch_bounds__(256) void norm_transpose_kernel(
    const float* __restrict__ X, unsigned short* __restrict__ Y,
    int C, int L, int doNorm)
{
    __shared__ float tile[512 * 17];      // [c][pos] padded: conflict-free
    __shared__ float red[16][17];
    __shared__ float meanS[16], scaleS[16];

    int b  = blockIdx.z;
    int l0 = blockIdx.x * 16;
    int tid = threadIdx.x;
    int px = tid & 15;                    // position in group
    int cg = tid >> 4;                    // channel group 0..15 (32 ch each)

    const float* Xb = X + (long long)b * C * L;
    for (int c = cg * 32; c < cg * 32 + 32; ++c)
        tile[c * 17 + px] = Xb[(long long)c * L + l0 + px];
    __syncthreads();

    float s = 0.f;
    for (int c = cg * 32; c < cg * 32 + 32; ++c) s += tile[c * 17 + px];
    red[px][cg] = s;
    __syncthreads();
    if (cg == 0) {
        float m = 0.f;
        for (int i = 0; i < 16; i++) m += red[px][i];
        meanS[px] = m / (float)C;
    }
    __syncthreads();
    float mean = meanS[px];
    float sq = 0.f;
    for (int c = cg * 32; c < cg * 32 + 32; ++c) {
        float d = tile[c * 17 + px] - mean;
        sq += d * d;
    }
    red[px][cg] = sq;
    __syncthreads();
    if (cg == 0) {
        float m = 0.f;
        for (int i = 0; i < 16; i++) m += red[px][i];
        scaleS[px] = 1.0f / (sqrtf(m) + EPSV);
    }
    __syncthreads();

    unsigned short* Yb = Y + ((long long)b * L + l0) * C;
    for (int idx = tid; idx < 16 * 512; idx += 256) {
        int p = idx >> 9, c = idx & 511;
        float x = tile[c * 17 + p];
        float val = doNorm ? (x - meanS[p]) * scaleS[p] : x;
        Yb[(long long)p * C + c] = f2bf_bits(val);
    }
}

// ---------------------------------------------------------------------------
// f32 -> bf16 convert (weights)
// ---------------------------------------------------------------------------
__global__ void f32_to_bf16_kernel(const float* __restrict__ src,
                                   unsigned short* __restrict__ dst, int n)
{
    int i = blockIdx.x * blockDim.x + threadIdx.x;
    if (i < n) dst[i] = f2bf_bits(src[i]);
}

// ---------------------------------------------------------------------------
// Generic bf16 WMMA GEMM:  D[row][col] = scale * sum_k A[row][k]*B[col][k]
// A: [rows][K] bf16 row-major (K contiguous).  B: [cols][K] bf16 (K contiguous).
// 128 threads = 4 waves; wave w computes rows [blockRow*64+w*16, +16) x 64 cols.
// biasMode: 0 none, 1 bias[col], 2 bias[row].
// outMode : 0 f32, 1 bf16, 2 f32 with (col >= lenMask[b]) -> NEGV.
// ---------------------------------------------------------------------------
__global__ __launch_bounds__(128) void gemm_bf16_wmma_kernel(
    const unsigned short* __restrict__ A, long long aStride,
    const unsigned short* __restrict__ Bm, long long bStride,
    void* __restrict__ Out, long long oStride, int ldc,
    int K, float scale,
    const float* __restrict__ bias, int biasMode,
    int outMode, const int* __restrict__ lenMask)
{
    const int bz   = blockIdx.z;
    const int lane = threadIdx.x & 31;
    const int wave = threadIdx.x >> 5;
    const int sub  = lane >> 4;       // 0 or 1
    const int r16  = lane & 15;
    const int rowBase = blockIdx.y * 64 + wave * 16;
    const int colBase = blockIdx.x * 64;

    const unsigned short* aRow = A + (long long)bz * aStride
                                   + (long long)(rowBase + r16) * K;
    const unsigned short* bB = Bm + (long long)bz * bStride;
    const unsigned short* bCol[4];
#pragma unroll
    for (int j = 0; j < 4; j++)
        bCol[j] = bB + (long long)(colBase + j * 16 + r16) * K;

    v8f acc[4] = {};

    for (int k0 = 0; k0 < K; k0 += 32) {
        union U { v16bf v; uint4 q[2]; } a, b0, b1, b2, b3;
        // A 16x32 bf16: lane<16 -> row r16, K[k0..k0+15];
        //               lane>=16 -> K[k0+16..k0+31]
        const int ka = k0 + sub * 16;
        a.q[0] = *(const uint4*)(aRow + ka);
        a.q[1] = *(const uint4*)(aRow + ka + 8);
        // B 32x16 bf16: lane<16 -> col r16, K[k0..k0+15]; lane>=16 -> K[k0+16..k0+31]
        const int kb = k0 + sub * 16;
        b0.q[0] = *(const uint4*)(bCol[0] + kb);
        b0.q[1] = *(const uint4*)(bCol[0] + kb + 8);
        b1.q[0] = *(const uint4*)(bCol[1] + kb);
        b1.q[1] = *(const uint4*)(bCol[1] + kb + 8);
        b2.q[0] = *(const uint4*)(bCol[2] + kb);
        b2.q[1] = *(const uint4*)(bCol[2] + kb + 8);
        b3.q[0] = *(const uint4*)(bCol[3] + kb);
        b3.q[1] = *(const uint4*)(bCol[3] + kb + 8);

        // speculative prefetch of next K-slab (global_prefetch_b8)
        __builtin_prefetch(aRow + ka + 32, 0, 3);
        __builtin_prefetch(bCol[0] + kb + 32, 0, 3);

        acc[0] = __builtin_amdgcn_wmma_f32_16x16x32_bf16(
                     false, a.v, false, b0.v, (short)0, acc[0], false, false);
        acc[1] = __builtin_amdgcn_wmma_f32_16x16x32_bf16(
                     false, a.v, false, b1.v, (short)0, acc[1], false, false);
        acc[2] = __builtin_amdgcn_wmma_f32_16x16x32_bf16(
                     false, a.v, false, b2.v, (short)0, acc[2], false, false);
        acc[3] = __builtin_amdgcn_wmma_f32_16x16x32_bf16(
                     false, a.v, false, b3.v, (short)0, acc[3], false, false);
    }

    // D layout: element i -> row = rowBase + i + 8*sub, col = colBase + j*16 + r16
#pragma unroll
    for (int j = 0; j < 4; j++) {
        const int col = colBase + j * 16 + r16;
#pragma unroll
        for (int i = 0; i < 8; i++) {
            const int row = rowBase + i + 8 * sub;
            float val = acc[j][i] * scale;
            if (biasMode == 1)      val += bias[col];
            else if (biasMode == 2) val += bias[row];
            const long long off = (long long)bz * oStride
                                + (long long)row * ldc + col;
            if (outMode == 0) {
                ((float*)Out)[off] = val;
            } else if (outMode == 1) {
                ((unsigned short*)Out)[off] = f2bf_bits(val);
            } else {
                if (col >= lenMask[bz]) val = NEGV;
                ((float*)Out)[off] = val;
            }
        }
    }
}

// ---------------------------------------------------------------------------
// Row softmax over M=2048 (256 threads, 8 elems/thread), in-place f32 +
// bf16 copy for the wg GEMM; zeros invalid rows and batch_alpha.
// ---------------------------------------------------------------------------
__global__ __launch_bounds__(256) void softmax_kernel(
    float* __restrict__ att, unsigned short* __restrict__ attbf,
    float* __restrict__ alpha, const int* __restrict__ lt, int N, int M)
{
    int b = blockIdx.z, n = blockIdx.x;
    long long ro = ((long long)b * N + n) * M;
    float* row = att + ro;
    int tid = threadIdx.x;

    float v[8];
    float mx = -3.4e38f;
#pragma unroll
    for (int i = 0; i < 8; i++) {
        v[i] = row[tid + i * 256];
        mx = fmaxf(mx, v[i]);
    }
    __shared__ float sm[8];
    for (int o = 16; o > 0; o >>= 1) mx = fmaxf(mx, __shfl_down(mx, o, 32));
    if ((tid & 31) == 0) sm[tid >> 5] = mx;
    __syncthreads();
    float bm = sm[0];
    for (int i = 1; i < 8; i++) bm = fmaxf(bm, sm[i]);
    __syncthreads();

    float s = 0.f;
#pragma unroll
    for (int i = 0; i < 8; i++) { v[i] = __expf(v[i] - bm); s += v[i]; }
    for (int o = 16; o > 0; o >>= 1) s += __shfl_down(s, o, 32);
    if ((tid & 31) == 0) sm[tid >> 5] = s;
    __syncthreads();
    float bs = 0.f;
    for (int i = 0; i < 8; i++) bs += sm[i];

    float rs = (n < lt[b]) ? (1.0f / bs) : 0.0f;   // rows beyond lt stay zero
#pragma unroll
    for (int i = 0; i < 8; i++) {
        float o = v[i] * rs;
        row[tid + i * 256] = o;
        attbf[ro + tid + i * 256] = f2bf_bits(o);
    }
    if (tid == 0) alpha[(long long)b * N + n] = 0.0f;
}

// ---------------------------------------------------------------------------
extern "C" void kernel_launch(void* const* d_in, const int* in_sizes, int n_in,
                              void* d_out, int out_size, void* d_ws, size_t ws_size,
                              hipStream_t stream)
{
    const int B = 8, C = 512, N = 2048, M = 2048, HW = 256 * 256;

    const float* xT  = (const float*)d_in[0];   // [B,C,N]
    const float* xR  = (const float*)d_in[1];   // [B,C,M]
    const float* rgb = (const float*)d_in[2];   // [B,C,M]
    const int*   lmT = (const int*)d_in[3];
    const int*   lmR = (const int*)d_in[4];
    const float* Wq = (const float*)d_in[8];
    const float* bq = (const float*)d_in[9];
    const float* Wk = (const float*)d_in[10];
    const float* bk = (const float*)d_in[11];
    const float* Wv = (const float*)d_in[12];
    const float* bv = (const float*)d_in[13];

    char* ws = (char*)d_ws;
    int* lt = (int*)ws;
    int* lr = (int*)(ws + 64);
    size_t off = 256;
    auto carve = [&](size_t bytes) -> void* {
        void* p = ws + off;
        off += (bytes + 255) & ~(size_t)255;
        return p;
    };
    unsigned short* xtb  = (unsigned short*)carve((size_t)B * N * C * 2); // [B][N][C]
    unsigned short* xrb  = (unsigned short*)carve((size_t)B * M * C * 2); // [B][M][C]
    unsigned short* rgbb = (unsigned short*)carve((size_t)B * M * C * 2); // [B][M][C]
    unsigned short* wqb  = (unsigned short*)carve((size_t)C * C * 2);
    unsigned short* wkb  = (unsigned short*)carve((size_t)C * C * 2);
    unsigned short* wvb  = (unsigned short*)carve((size_t)C * C * 2);
    unsigned short* qb   = (unsigned short*)carve((size_t)B * N * C * 2); // [B][N][C]
    unsigned short* kb   = (unsigned short*)carve((size_t)B * M * C * 2); // [B][M][C]
    unsigned short* vb   = (unsigned short*)carve((size_t)B * C * M * 2); // [B][C][M]
    unsigned short* attb = (unsigned short*)carve((size_t)B * N * M * 2); // [B][N][M]

    float* outWg    = (float*)d_out;                       // [B][C][N]
    float* outAtt   = outWg + (size_t)B * C * N;           // [B][N][M]
    float* outAlpha = outAtt + (size_t)B * N * M;          // [B][N]

    // 0) lengths
    hipMemsetAsync(d_ws, 0, 256, stream);
    len_reduce_kernel<<<dim3(B * 128), 256, 0, stream>>>(lmT, lmR, lt, lr, HW);

    // 1) normalize + transpose to bf16
    norm_transpose_kernel<<<dim3(N / 16, 1, B), 256, 0, stream>>>(xT,  xtb,  C, N, 1);
    norm_transpose_kernel<<<dim3(M / 16, 1, B), 256, 0, stream>>>(xR,  xrb,  C, M, 1);
    norm_transpose_kernel<<<dim3(M / 16, 1, B), 256, 0, stream>>>(rgb, rgbb, C, M, 0);

    // 2) weights -> bf16
    f32_to_bf16_kernel<<<dim3((C * C + 255) / 256), 256, 0, stream>>>(Wq, wqb, C * C);
    f32_to_bf16_kernel<<<dim3((C * C + 255) / 256), 256, 0, stream>>>(Wk, wkb, C * C);
    f32_to_bf16_kernel<<<dim3((C * C + 255) / 256), 256, 0, stream>>>(Wv, wvb, C * C);

    // 3) q = xtb @ Wq^T + bq   -> qb [B][N][C] bf16
    gemm_bf16_wmma_kernel<<<dim3(C / 64, N / 64, B), 128, 0, stream>>>(
        xtb, (long long)N * C, wqb, 0, qb, (long long)N * C, C,
        C, 1.0f, bq, 1, 1, nullptr);
    // 4) k = xrb @ Wk^T + bk   -> kb [B][M][C] bf16
    gemm_bf16_wmma_kernel<<<dim3(C / 64, M / 64, B), 128, 0, stream>>>(
        xrb, (long long)M * C, wkb, 0, kb, (long long)M * C, C,
        C, 1.0f, bk, 1, 1, nullptr);
    // 5) v = Wv @ rgb + bv     -> vb [B][C][M] bf16 (bias on rows = out channel)
    gemm_bf16_wmma_kernel<<<dim3(M / 64, C / 64, B), 128, 0, stream>>>(
        wvb, 0, rgbb, (long long)M * C, vb, (long long)C * M, M,
        C, 1.0f, bv, 2, 1, nullptr);
    // 6) sim = (q . k) / TEMP, ref-masked -> outAtt f32 [B][N][M]
    gemm_bf16_wmma_kernel<<<dim3(M / 64, N / 64, B), 128, 0, stream>>>(
        qb, (long long)N * C, kb, (long long)M * C, outAtt, (long long)N * M, M,
        C, 100.0f, nullptr, 0, 2, lr);
    // 7) softmax rows, target-row mask, bf16 copy, alpha=0
    softmax_kernel<<<dim3(N, 1, B), 256, 0, stream>>>(outAtt, attb, outAlpha, lt, N, M);
    // 8) wg_transp = v @ att^T -> outWg f32 [B][C][N]
    gemm_bf16_wmma_kernel<<<dim3(N / 64, C / 64, B), 128, 0, stream>>>(
        vb, (long long)C * M, attb, (long long)N * M, outWg, (long long)C * N, N,
        M, 1.0f, nullptr, 0, 0, nullptr);
}
